// AMPBERT_20383914786865
// MI455X (gfx1250) — compile-verified, matrix operand order checked by reading, not run on hardware
//
#include <hip/hip_runtime.h>
#include <math.h>

// ---------------------------------------------------------------------------
// Model constants (from reference)
// ---------------------------------------------------------------------------
constexpr int BB = 8, TT = 256, NMSK = 40, VV = 30;
constexpr int NE = 1024, DMODEL = 1024, DK = 64, DV = 64, HH = 16, DFFC = 4096, LL = 8;
constexpr int BT = BB * TT;
constexpr float EPSLN = 1e-5f;

typedef __attribute__((ext_vector_type(16))) __bf16 v16bf;
typedef __attribute__((ext_vector_type(8)))  float  v8f;
typedef __attribute__((ext_vector_type(4)))  int    v4i;

union FragU { v16bf v; unsigned int u[8]; };
union V8U  { uint4 q; __bf16 h[8]; };

// --- CDNA5 async global->LDS path (guarded; falls back to reg staging) -----
#if defined(__has_builtin)
#  if __has_builtin(__builtin_amdgcn_global_load_async_to_lds_b128)
#    define USE_ASYNC_LDS 1
#  endif
#endif
#ifndef USE_ASYNC_LDS
#  define USE_ASYNC_LDS 0
#endif

#if USE_ASYNC_LDS
#  define GLOBAL_TO_LDS_B128(gp, lp)                                          \
      __builtin_amdgcn_global_load_async_to_lds_b128((v4i*)(gp), (v4i*)(lp), 0, 0)
#  if __has_builtin(__builtin_amdgcn_s_wait_asynccnt)
#    define WAIT_ASYNC() __builtin_amdgcn_s_wait_asynccnt(0)
#  else
#    define WAIT_ASYNC() asm volatile("s_wait_asynccnt 0" ::: "memory")
#  endif
#else
#  define WAIT_ASYNC()
#endif

// ---------------------------------------------------------------------------
// Block reductions (wave32-aware: 256-thread block = 8 waves)
// ---------------------------------------------------------------------------
__device__ __forceinline__ float blockReduceSum(float v) {
    __shared__ float red[8];
    int lane = threadIdx.x & 31, wv = threadIdx.x >> 5;
    #pragma unroll
    for (int o = 16; o > 0; o >>= 1) v += __shfl_xor(v, o, 32);
    if (lane == 0) red[wv] = v;
    __syncthreads();
    if (wv == 0) {
        float t = (lane < 8) ? red[lane] : 0.0f;
        #pragma unroll
        for (int o = 4; o > 0; o >>= 1) t += __shfl_xor(t, o, 32);
        if (lane == 0) red[0] = t;
    }
    __syncthreads();
    float r = red[0];
    __syncthreads();
    return r;
}

__device__ __forceinline__ float blockReduceMax(float v) {
    __shared__ float red[8];
    int lane = threadIdx.x & 31, wv = threadIdx.x >> 5;
    #pragma unroll
    for (int o = 16; o > 0; o >>= 1) v = fmaxf(v, __shfl_xor(v, o, 32));
    if (lane == 0) red[wv] = v;
    __syncthreads();
    if (wv == 0) {
        float t = (lane < 8) ? red[lane] : -3.0e38f;
        #pragma unroll
        for (int o = 4; o > 0; o >>= 1) t = fmaxf(t, __shfl_xor(t, o, 32));
        if (lane == 0) red[0] = t;
    }
    __syncthreads();
    float r = red[0];
    __syncthreads();
    return r;
}

// ---------------------------------------------------------------------------
// Generic batched WMMA GEMM:  C = epi( scale*(A@B) + bias [+ res] )
//   A: bf16 [M,K] row-major (lda); B: bf16 [K,N] (ldb) or [N,K] if transB.
//   C: f32 or bf16 (ldc). Batch z -> (zb = z/bh, zh = z%bh) split strides.
// Block tile 128x128, 8 waves = 4(M) x 2(N), wave tile 32x64 = 2x4 WMMA
// 16x16 tiles, K chunks of 32, double-buffered LDS, async global->LDS fills.
// OOB rows/cols are CLAMPED on load (A row r only affects C row r, B col n
// only affects C col n; both are discarded by the epilogue bounds checks).
// ---------------------------------------------------------------------------
enum { EPI_NONE = 0, EPI_BIAS = 1, EPI_BIAS_GELU = 2, EPI_SCALE_MASK = 3, EPI_BIAS_RES = 4 };

template <int EPI, bool OUTBF>
__global__ __launch_bounds__(256)
void gemm_wmma(const __bf16* __restrict__ A, long sAb, long sAh, int lda,
               const __bf16* __restrict__ Bm, long sBb, long sBh, int ldb, int transB,
               void* __restrict__ Cv, long sCb, long sCh, int ldc,
               const float* __restrict__ bias, long sBias,
               const float* __restrict__ res,
               float scale, int M, int N, int K, int bh) {
    constexpr int BM = 128, BN = 128, BK = 32;
    constexpr int LDSP = BK + 8;  // padded LDS row stride (elements)
    __shared__ __align__(16) __bf16 As[2][BM * LDSP];
    __shared__ __align__(16) __bf16 Bs[2][BN * LDSP];

    const int tid  = threadIdx.x;
    const int lane = tid & 31;
    const int wv   = tid >> 5;
    const int wm   = wv & 3;    // wave row (0..3): 32 rows each
    const int wn   = wv >> 2;   // wave col (0..1): 64 cols each
    const int m0   = blockIdx.y * BM;
    const int n0   = blockIdx.x * BN;
    const int zb   = blockIdx.z / bh;
    const int zh   = blockIdx.z % bh;

    A  += (long)zb * sAb + (long)zh * sAh;
    Bm += (long)zb * sBb + (long)zh * sBh;
    const long cOff = (long)zb * sCb + (long)zh * sCh;

    // vectorized staging decomposition: 128x32 tile = 512 x (8 bf16); 2 per thread
    const int stR = tid >> 2;        // 0..63 (+64 on second pass)
    const int stC = (tid & 3) << 3;  // 0,8,16,24
    // scalar-gather staging for B=[K,N] (activations): one n per thread, 16 k's
    const int bn  = tid & 127;
    const int bk0 = (tid >> 7) << 4; // 0 or 16

    V8U aReg[2], bReg[2];
    (void)aReg;

    // issue stage for chunk k0 into LDS buffer `buf`
    auto stage = [&](int k0, int buf) {
#if USE_ASYNC_LDS
        #pragma unroll
        for (int it = 0; it < 2; ++it) {
            int gr = m0 + stR + it * 64; if (gr >= M) gr = M - 1;  // clamp
            GLOBAL_TO_LDS_B128(A + (long)gr * lda + k0 + stC,
                               &As[buf][(stR + it * 64) * LDSP + stC]);
        }
        if (transB) {
            #pragma unroll
            for (int it = 0; it < 2; ++it) {
                int gn = n0 + stR + it * 64; if (gn >= N) gn = N - 1;  // clamp
                GLOBAL_TO_LDS_B128(Bm + (long)gn * ldb + k0 + stC,
                                   &Bs[buf][(stR + it * 64) * LDSP + stC]);
            }
        } else {
            int gn = n0 + bn;
            #pragma unroll
            for (int it = 0; it < 2; ++it)
                #pragma unroll
                for (int i = 0; i < 8; ++i)
                    bReg[it].h[i] = (gn < N) ? Bm[(long)(k0 + bk0 + it * 8 + i) * ldb + gn]
                                             : (__bf16)0.0f;
        }
#else
        (void)buf;
        #pragma unroll
        for (int it = 0; it < 2; ++it) {
            int gr = m0 + stR + it * 64;
            if (gr < M) aReg[it].q = *reinterpret_cast<const uint4*>(A + (long)gr * lda + k0 + stC);
            else        aReg[it].q = make_uint4(0u, 0u, 0u, 0u);
        }
        if (transB) {
            #pragma unroll
            for (int it = 0; it < 2; ++it) {
                int gn = n0 + stR + it * 64;
                if (gn < N) bReg[it].q = *reinterpret_cast<const uint4*>(Bm + (long)gn * ldb + k0 + stC);
                else        bReg[it].q = make_uint4(0u, 0u, 0u, 0u);
            }
        } else {
            int gn = n0 + bn;
            #pragma unroll
            for (int it = 0; it < 2; ++it)
                #pragma unroll
                for (int i = 0; i < 8; ++i)
                    bReg[it].h[i] = (gn < N) ? Bm[(long)(k0 + bk0 + it * 8 + i) * ldb + gn]
                                             : (__bf16)0.0f;
        }
#endif
    };
    // flush any register-staged data into LDS buffer `buf`
    auto flush = [&](int buf) {
#if USE_ASYNC_LDS
        if (!transB) {
            #pragma unroll
            for (int it = 0; it < 2; ++it)
                *reinterpret_cast<uint4*>(&Bs[buf][bn * LDSP + bk0 + it * 8]) = bReg[it].q;
        }
#else
        #pragma unroll
        for (int it = 0; it < 2; ++it)
            *reinterpret_cast<uint4*>(&As[buf][(stR + it * 64) * LDSP + stC]) = aReg[it].q;
        if (transB) {
            #pragma unroll
            for (int it = 0; it < 2; ++it)
                *reinterpret_cast<uint4*>(&Bs[buf][(stR + it * 64) * LDSP + stC]) = bReg[it].q;
        } else {
            #pragma unroll
            for (int it = 0; it < 2; ++it)
                *reinterpret_cast<uint4*>(&Bs[buf][bn * LDSP + bk0 + it * 8]) = bReg[it].q;
        }
#endif
    };

    v8f acc[2][4] = {};
    const int row16 = lane & 15;
    const int khalf = (lane >> 4) << 3;  // lanes 16..31 take the +8 K half

    const int nk = K / BK;
    stage(0, 0);
    flush(0);
    WAIT_ASYNC();
    __syncthreads();

    for (int kc = 0; kc < nk; ++kc) {
        const int buf = kc & 1;
        if (kc + 1 < nk) stage((kc + 1) * BK, buf ^ 1);  // fills overlap compute

        FragU fa[2], fb[4];
        #pragma unroll
        for (int r = 0; r < 2; ++r) {
            int rr = wm * 32 + r * 16 + row16;
            #pragma unroll
            for (int g = 0; g < 8; ++g) {
                int kk = ((g < 4) ? (g << 1) : (16 + ((g - 4) << 1))) + khalf;
                fa[r].u[g] = *reinterpret_cast<const unsigned int*>(&As[buf][rr * LDSP + kk]);
            }
        }
        #pragma unroll
        for (int c = 0; c < 4; ++c) {
            int nn = wn * 64 + c * 16 + row16;
            #pragma unroll
            for (int g = 0; g < 8; ++g) {
                int kk = ((g < 4) ? (g << 1) : (16 + ((g - 4) << 1))) + khalf;
                fb[c].u[g] = *reinterpret_cast<const unsigned int*>(&Bs[buf][nn * LDSP + kk]);
            }
        }
        #pragma unroll
        for (int r = 0; r < 2; ++r)
            #pragma unroll
            for (int c = 0; c < 4; ++c)
                acc[r][c] = __builtin_amdgcn_wmma_f32_16x16x32_bf16(
                    false, fa[r].v, false, fb[c].v, (short)0, acc[r][c], false, false);

        if (kc + 1 < nk) {
            flush(buf ^ 1);   // safe: barrier at end of prev iter fenced readers
            WAIT_ASYNC();
            __syncthreads();
        }
    }

    // epilogue: VGPR j -> lanes 0-15: M=j, lanes 16-31: M=8+j; N = lane&15
    float*  Cf = reinterpret_cast<float*>(Cv);
    __bf16* Cb = reinterpret_cast<__bf16*>(Cv);
    #pragma unroll
    for (int r = 0; r < 2; ++r) {
        #pragma unroll
        for (int c = 0; c < 4; ++c) {
            int colg  = n0 + wn * 64 + c * 16 + row16;
            int rbase = m0 + wm * 32 + r * 16 + ((lane >> 4) << 3);
            if (colg >= N) continue;
            #pragma unroll
            for (int j = 0; j < 8; ++j) {
                int rowg = rbase + j;
                if (rowg >= M) continue;
                float v = acc[r][c][j];
                if (EPI == EPI_SCALE_MASK) v = v * scale + bias[(long)zb * sBias + colg];
                if (EPI == EPI_BIAS || EPI == EPI_BIAS_GELU || EPI == EPI_BIAS_RES) v += bias[colg];
                if (EPI == EPI_BIAS_RES)  v += res[(long)rowg * ldc + colg];
                if (EPI == EPI_BIAS_GELU) v = 0.5f * v * (1.0f + erff(v * 0.70710678118f));
                long ci = cOff + (long)rowg * ldc + colg;
                if (OUTBF) Cb[ci] = (__bf16)v; else Cf[ci] = v;
            }
        }
    }
}

template <int EPI, bool OUTBF>
static inline void gemmL(hipStream_t st,
                         const __bf16* A, long sAb, long sAh, int lda,
                         const __bf16* Bm, long sBb, long sBh, int ldb, int transB,
                         void* C, long sCb, long sCh, int ldc,
                         const float* bias, long sBias, const float* res,
                         float scale, int M, int N, int K, int batches, int bh) {
    dim3 grid((unsigned)((N + 127) / 128), (unsigned)((M + 127) / 128), (unsigned)batches);
    gemm_wmma<EPI, OUTBF><<<grid, 256, 0, st>>>(A, sAb, sAh, lda, Bm, sBb, sBh, ldb, transB,
                                                C, sCb, sCh, ldc, bias, sBias, res,
                                                scale, M, N, K, bh);
}

// ---------------------------------------------------------------------------
// Weight cast + transpose: in [z][K][N] f32 -> out [z][N][K] bf16 (32x32 tiles)
// ---------------------------------------------------------------------------
__global__ __launch_bounds__(256)
void convT_kernel(const float* __restrict__ in, __bf16* __restrict__ out, int K, int N) {
    __shared__ float tile[32][33];
    const long base = (long)blockIdx.z * (long)K * N;
    const int n0 = blockIdx.x * 32, k0 = blockIdx.y * 32;
    const int tx = threadIdx.x & 31, ty = threadIdx.x >> 5;
    #pragma unroll
    for (int i = 0; i < 32; i += 8) {
        int k = k0 + ty + i, n = n0 + tx;
        tile[ty + i][tx] = (k < K && n < N) ? in[base + (long)k * N + n] : 0.0f;
    }
    __syncthreads();
    #pragma unroll
    for (int i = 0; i < 32; i += 8) {
        int n = n0 + ty + i, k = k0 + tx;
        if (n < N && k < K) out[base + (long)n * K + k] = (__bf16)tile[tx][ty + i];
    }
}

// ---------------------------------------------------------------------------
// Elementwise / normalization / softmax / gather / loss kernels
// ---------------------------------------------------------------------------
__global__ void mask_kernel(const int* __restrict__ seqs, float* __restrict__ mb) {
    int i = blockIdx.x * 256 + threadIdx.x;
    if (i < BT) mb[i] = (seqs[i] == 0) ? -1e9f : 0.0f;
}

// emb = tok_embed[seq] + pos_embed ; out = bf16(LN(emb))
__global__ __launch_bounds__(256)
void embed_ln_kernel(const int* __restrict__ seqs, const float* __restrict__ tok,
                     const float* __restrict__ pos, const float* __restrict__ g,
                     const float* __restrict__ bta, __bf16* __restrict__ outb) {
    int row = blockIdx.x;            // 0..BT-1
    int t   = row & (TT - 1);
    int tokid = seqs[row];
    float e[4]; float s = 0.0f;
    #pragma unroll
    for (int i = 0; i < 4; ++i) {
        int c = threadIdx.x + i * 256;
        e[i] = tok[(long)tokid * NE + c] + pos[(long)t * NE + c];
        s += e[i];
    }
    float mean = blockReduceSum(s) * (1.0f / NE);
    float vs = 0.0f;
    #pragma unroll
    for (int i = 0; i < 4; ++i) { float d = e[i] - mean; vs += d * d; }
    float rstd = rsqrtf(blockReduceSum(vs) * (1.0f / NE) + EPSLN);
    #pragma unroll
    for (int i = 0; i < 4; ++i) {
        int c = threadIdx.x + i * 256;
        outb[(long)row * NE + c] = (__bf16)((e[i] - mean) * rstd * g[c] + bta[c]);
    }
}

// LN over D=1024 rows; bf16 out always, f32 out optional (residual keep)
__global__ __launch_bounds__(256)
void ln_kernel(const float* __restrict__ x, const float* __restrict__ g,
               const float* __restrict__ bta, __bf16* __restrict__ outb,
               float* __restrict__ outf) {
    int row = blockIdx.x;
    const float* xr = x + (long)row * DMODEL;
    float e[4]; float s = 0.0f;
    #pragma unroll
    for (int i = 0; i < 4; ++i) { e[i] = xr[threadIdx.x + i * 256]; s += e[i]; }
    float mean = blockReduceSum(s) * (1.0f / DMODEL);
    float vs = 0.0f;
    #pragma unroll
    for (int i = 0; i < 4; ++i) { float d = e[i] - mean; vs += d * d; }
    float rstd = rsqrtf(blockReduceSum(vs) * (1.0f / DMODEL) + EPSLN);
    #pragma unroll
    for (int i = 0; i < 4; ++i) {
        int c = threadIdx.x + i * 256;
        float o = (e[i] - mean) * rstd * g[c] + bta[c];
        outb[(long)row * DMODEL + c] = (__bf16)o;
        if (outf) outf[(long)row * DMODEL + c] = o;
    }
}

// softmax over rows of length TT=256 (mask already folded in as -1e9), bf16 out
__global__ __launch_bounds__(256)
void softmax_kernel(const float* __restrict__ s, __bf16* __restrict__ out) {
    long row = blockIdx.x;
    float v  = s[row * TT + threadIdx.x];
    float mx = blockReduceMax(v);
    float e  = __expf(v - mx);
    float sum = blockReduceSum(e);
    out[row * TT + threadIdx.x] = (__bf16)(e / sum);
}

__global__ __launch_bounds__(256)
void gather_kernel(const __bf16* __restrict__ x, const int* __restrict__ pos,
                   __bf16* __restrict__ out) {
    int row = blockIdx.x;            // 0..B*NM-1
    int b   = row / NMSK;
    int p   = pos[row];
    const __bf16* src = x + ((long)(b * TT + p)) * DMODEL;
    __bf16* dst = out + (long)row * DMODEL;
    for (int i = threadIdx.x; i < DMODEL; i += 256) dst[i] = src[i];
}

// per-row -log_softmax(logits)[label]  (one wave per row; V=30 <= 32 lanes)
__global__ __launch_bounds__(32)
void loss_rows_kernel(const float* __restrict__ logits, const int* __restrict__ labels,
                      float* __restrict__ rowloss) {
    int row = blockIdx.x, lane = threadIdx.x;
    float v = (lane < VV) ? logits[(long)row * VV + lane] : -3.0e38f;
    float mx = v;
    #pragma unroll
    for (int o = 16; o > 0; o >>= 1) mx = fmaxf(mx, __shfl_xor(mx, o, 32));
    float e = (lane < VV) ? __expf(v - mx) : 0.0f;
    float s = e;
    #pragma unroll
    for (int o = 16; o > 0; o >>= 1) s += __shfl_xor(s, o, 32);
    if (lane == 0) {
        int lbl = labels[row];
        float lp = logits[(long)row * VV + lbl] - mx - __logf(s);
        rowloss[row] = -lp;
    }
}

// deterministic fixed-order reduce (no float atomics)
__global__ void loss_reduce_kernel(const float* __restrict__ rowloss, float* __restrict__ out) {
    if (threadIdx.x == 0 && blockIdx.x == 0) {
        float s = 0.0f;
        for (int i = 0; i < BB * NMSK; ++i) s += rowloss[i];
        *out = s / (float)(BB * NMSK);
    }
}

// ---------------------------------------------------------------------------
// Orchestration
// ---------------------------------------------------------------------------
extern "C" void kernel_launch(void* const* d_in, const int* in_sizes, int n_in,
                              void* d_out, int out_size, void* d_ws, size_t ws_size,
                              hipStream_t stream) {
    (void)in_sizes; (void)n_in; (void)out_size; (void)ws_size;

    const int*   seqs   = (const int*)d_in[0];
    const int*   mpos   = (const int*)d_in[1];
    const int*   mtok   = (const int*)d_in[2];
    const float* tok_e  = (const float*)d_in[3];
    const float* pos_e  = (const float*)d_in[4];
    const float* norm_g = (const float*)d_in[5];
    const float* norm_b = (const float*)d_in[6];
    const float* fc_w   = (const float*)d_in[7];
    const float* fc_b   = (const float*)d_in[8];
    const float* ln1_g  = (const float*)d_in[9];
    const float* ln1_b  = (const float*)d_in[10];
    const float* wq_w   = (const float*)d_in[11];
    const float* wq_b   = (const float*)d_in[12];
    const float* wk_w   = (const float*)d_in[13];
    const float* wk_b   = (const float*)d_in[14];
    const float* wv_w   = (const float*)d_in[15];
    const float* wv_b   = (const float*)d_in[16];
    const float* ao_w   = (const float*)d_in[17];
    const float* ao_b   = (const float*)d_in[18];
    const float* ln2_g  = (const float*)d_in[19];
    const float* ln2_b  = (const float*)d_in[20];
    const float* f1_w   = (const float*)d_in[21];
    const float* f1_b   = (const float*)d_in[22];
    const float* f2_w   = (const float*)d_in[23];
    const float* f2_b   = (const float*)d_in[24];
    const float* lnf_g  = (const float*)d_in[25];
    const float* lnf_b  = (const float*)d_in[26];
    const float* lin_w  = (const float*)d_in[27];
    const float* lin_b  = (const float*)d_in[28];
    const float* head_w = (const float*)d_in[29];

    // ---- workspace carve (256B aligned) ----
    char* p = (char*)d_ws;
    auto carve = [&](size_t elems, size_t esize) -> void* {
        void* r = (void*)p;
        size_t bytes = (elems * esize + 255) & ~(size_t)255;
        p += bytes;
        return r;
    };
    // transposed bf16 weights: [N,K] layout
    __bf16* fcw   = (__bf16*)carve((size_t)NE * DMODEL, 2);
    __bf16* wqw   = (__bf16*)carve((size_t)LL * DMODEL * HH * DK, 2);
    __bf16* wkw   = (__bf16*)carve((size_t)LL * DMODEL * HH * DK, 2);
    __bf16* wvw   = (__bf16*)carve((size_t)LL * DMODEL * HH * DV, 2);
    __bf16* aow   = (__bf16*)carve((size_t)LL * HH * DV * DMODEL, 2);
    __bf16* f1w   = (__bf16*)carve((size_t)LL * DMODEL * DFFC, 2);
    __bf16* f2w   = (__bf16*)carve((size_t)LL * DFFC * DMODEL, 2);
    __bf16* linw  = (__bf16*)carve((size_t)DMODEL * DMODEL, 2);
    __bf16* headw = (__bf16*)carve((size_t)DMODEL * VV, 2);
    float*  xF      = (float*)carve((size_t)BT * DMODEL, 4);
    float*  hF      = (float*)carve((size_t)BT * DMODEL, 4);
    float*  scoresF = (float*)carve((size_t)BB * HH * TT * TT, 4);
    float*  maskb   = (float*)carve((size_t)BT, 4);
    float*  rowloss = (float*)carve((size_t)BB * NMSK, 4);
    __bf16* aB    = (__bf16*)carve((size_t)BT * DMODEL, 2);
    __bf16* qB    = (__bf16*)carve((size_t)BT * DMODEL, 2);
    __bf16* kB    = (__bf16*)carve((size_t)BT * DMODEL, 2);
    __bf16* vB    = (__bf16*)carve((size_t)BT * DMODEL, 2);
    __bf16* attnB = (__bf16*)carve((size_t)BB * HH * TT * TT, 2);
    __bf16* ctxB  = (__bf16*)carve((size_t)BT * DMODEL, 2);
    __bf16* midB  = (__bf16*)carve((size_t)BT * DFFC, 2);
    __bf16* gB    = (__bf16*)carve((size_t)BB * NMSK * DMODEL, 2);
    __bf16* linoB = (__bf16*)carve((size_t)BB * NMSK * DMODEL, 2);

    // ---- one-time bf16 cast + transpose of all weights ([K,N] -> [N,K]) ----
    auto convT = [&](const float* src, __bf16* dst, int Kd, int Nd, int batches) {
        dim3 g((unsigned)((Nd + 31) / 32), (unsigned)((Kd + 31) / 32), (unsigned)batches);
        convT_kernel<<<g, 256, 0, stream>>>(src, dst, Kd, Nd);
    };
    convT(fc_w,   fcw,   NE, DMODEL, 1);
    convT(wq_w,   wqw,   DMODEL, DMODEL, LL);
    convT(wk_w,   wkw,   DMODEL, DMODEL, LL);
    convT(wv_w,   wvw,   DMODEL, DMODEL, LL);
    convT(ao_w,   aow,   DMODEL, DMODEL, LL);
    convT(f1_w,   f1w,   DMODEL, DFFC,   LL);
    convT(f2_w,   f2w,   DFFC,   DMODEL, LL);
    convT(lin_w,  linw,  DMODEL, DMODEL, 1);
    convT(head_w, headw, DMODEL, VV,     1);

    mask_kernel<<<dim3((BT + 255) / 256), 256, 0, stream>>>(seqs, maskb);
    embed_ln_kernel<<<dim3(BT), 256, 0, stream>>>(seqs, tok_e, pos_e, norm_g, norm_b, aB);

    // x = LN(emb) @ fc_w + fc_b                                  [BT, DM] f32
    gemmL<EPI_BIAS, false>(stream, aB, 0, 0, NE, fcw, 0, 0, NE, 1,
                           xF, 0, 0, DMODEL, fc_b, 0, nullptr, 1.0f,
                           BT, DMODEL, NE, 1, 1);

    const long W1 = (long)DMODEL * DMODEL;
    for (int l = 0; l < LL; ++l) {
        // h = LN1(x): bf16 for GEMMs + f32 kept as attention residual
        ln_kernel<<<dim3(BT), 256, 0, stream>>>(xF, ln1_g + (long)l * DMODEL,
                                                ln1_b + (long)l * DMODEL, aB, hF);
        // Q,K,V projections (bf16 out); weights are [N,K] -> transB fast path
        gemmL<EPI_BIAS, true>(stream, aB, 0, 0, DMODEL, wqw + l * W1, 0, 0, DMODEL, 1,
                              qB, 0, 0, DMODEL, wq_b + (long)l * DMODEL, 0, nullptr, 1.0f,
                              BT, DMODEL, DMODEL, 1, 1);
        gemmL<EPI_BIAS, true>(stream, aB, 0, 0, DMODEL, wkw + l * W1, 0, 0, DMODEL, 1,
                              kB, 0, 0, DMODEL, wk_b + (long)l * DMODEL, 0, nullptr, 1.0f,
                              BT, DMODEL, DMODEL, 1, 1);
        gemmL<EPI_BIAS, true>(stream, aB, 0, 0, DMODEL, wvw + l * W1, 0, 0, DMODEL, 1,
                              vB, 0, 0, DMODEL, wv_b + (long)l * DMODEL, 0, nullptr, 1.0f,
                              BT, DMODEL, DMODEL, 1, 1);
        // scores = Q @ K^T * 1/sqrt(DK) + pad_mask   (batched over B*H)
        gemmL<EPI_SCALE_MASK, false>(stream,
            qB, (long)TT * DMODEL, DK, DMODEL,
            kB, (long)TT * DMODEL, DK, DMODEL, 1 /* K is [T,DK] = B^T */,
            scoresF, (long)HH * TT * TT, (long)TT * TT, TT,
            maskb, TT, nullptr, 0.125f, TT, TT, DK, BB * HH, HH);
        softmax_kernel<<<dim3(BB * HH * TT), 256, 0, stream>>>(scoresF, attnB);
        // ctx = attn @ V   (batched, bf16 out interleaved to [B,T,H*DV])
        gemmL<EPI_NONE, true>(stream,
            attnB, (long)HH * TT * TT, (long)TT * TT, TT,
            vB, (long)TT * DMODEL, DV, DMODEL, 0,
            ctxB, (long)TT * DMODEL, DV, DMODEL,
            nullptr, 0, nullptr, 1.0f, TT, DV, TT, BB * HH, HH);
        // out = ctx @ ao_w + ao_b + h (residual = post-LN1 h)
        gemmL<EPI_BIAS_RES, false>(stream, ctxB, 0, 0, DMODEL, aow + l * W1, 0, 0, DMODEL, 1,
                                   xF, 0, 0, DMODEL, ao_b + (long)l * DMODEL, 0, hF, 1.0f,
                                   BT, DMODEL, DMODEL, 1, 1);
        // LN2 then FFN: gelu(x@w1+b1)@w2+b2  (no residual per reference)
        ln_kernel<<<dim3(BT), 256, 0, stream>>>(xF, ln2_g + (long)l * DMODEL,
                                                ln2_b + (long)l * DMODEL, aB, nullptr);
        gemmL<EPI_BIAS_GELU, true>(stream, aB, 0, 0, DMODEL,
                                   f1w + (long)l * DMODEL * DFFC, 0, 0, DMODEL, 1,
                                   midB, 0, 0, DFFC, f1_b + (long)l * DFFC, 0, nullptr, 1.0f,
                                   BT, DFFC, DMODEL, 1, 1);
        gemmL<EPI_BIAS, false>(stream, midB, 0, 0, DFFC,
                               f2w + (long)l * DFFC * DMODEL, 0, 0, DFFC, 1,
                               xF, 0, 0, DMODEL, f2_b + (long)l * DMODEL, 0, nullptr, 1.0f,
                               BT, DMODEL, DFFC, 1, 1);
    }

    // final LN -> gather masked positions -> gelu(h@lin+b) -> logits -> loss
    ln_kernel<<<dim3(BT), 256, 0, stream>>>(xF, lnf_g, lnf_b, aB, nullptr);
    gather_kernel<<<dim3(BB * NMSK), 256, 0, stream>>>(aB, mpos, gB);
    gemmL<EPI_BIAS_GELU, true>(stream, gB, 0, 0, DMODEL, linw, 0, 0, DMODEL, 1,
                               linoB, 0, 0, DMODEL, lin_b, 0, nullptr, 1.0f,
                               BB * NMSK, DMODEL, DMODEL, 1, 1);
    float* logits = (float*)d_out;
    gemmL<EPI_NONE, false>(stream, linoB, 0, 0, DMODEL, headw, 0, 0, DMODEL, 1,
                           logits, 0, 0, VV, nullptr, 0, nullptr, 1.0f,
                           BB * NMSK, VV, DMODEL, 1, 1);
    loss_rows_kernel<<<dim3(BB * NMSK), 32, 0, stream>>>(logits, mtok, rowloss);
    loss_reduce_kernel<<<dim3(1), 32, 0, stream>>>(rowloss, logits + (long)BB * NMSK * VV);
}
